// SparseMoELanguageModel_39067022525030
// MI455X (gfx1250) — matrix operand order repaired and checked by reading, not compiled, not run on hardware
//
#include <hip/hip_runtime.h>
#include <hip/hip_bf16.h>
#include <math.h>

// ---- model dims ----
#define Bc   4
#define Tc   1024
#define Cc   1024
#define Hc   16
#define HDc  64
#define Lc   2
#define Ec   8
#define Nc   (Bc*Tc)      // 4096 tokens
#define CAPc 1024
#define C3   (3*Cc)       // 3072
#define C4   (4*Cc)       // 4096

typedef __attribute__((ext_vector_type(16))) __bf16 v16bf;
typedef __attribute__((ext_vector_type(8)))  float  v8f;
typedef __attribute__((ext_vector_type(2)))  __bf16 bf16x2;
typedef __attribute__((ext_vector_type(4)))  unsigned int v4u;
typedef __attribute__((ext_vector_type(8)))  int v8i;
typedef __attribute__((ext_vector_type(4)))  int v4i;

#if defined(__has_builtin)
#if __has_builtin(__builtin_amdgcn_tensor_load_to_lds) && __has_builtin(__builtin_amdgcn_s_wait_tensorcnt)
#define USE_TDM 1
#else
#define USE_TDM 0
#endif
#else
#define USE_TDM 0
#endif

union FragB { v16bf v; uint4 u[2]; };

__device__ inline unsigned pk_bf16(float a, float b) {
    bf16x2 h;
    h[0] = (__bf16)a;
    h[1] = (__bf16)b;
    return __builtin_bit_cast(unsigned, h);   // -> v_cvt_pk_bf16_f32
}

// ---------------- embedding: X = tok_emb[ids] + pos_emb ----------------
__global__ void k_embed(const int* __restrict__ ids, const float* __restrict__ tok,
                        const float* __restrict__ pos, float* __restrict__ X) {
    int g = blockIdx.x * 256 + threadIdx.x;       // Nc*Cc threads
    int row = g >> 10, c = g & (Cc - 1);
    int t = row & (Tc - 1);
    X[g] = tok[(size_t)ids[row] * Cc + c] + pos[(size_t)t * Cc + c];
}

// ---------------- layernorm: writes f32 and bf16 ----------------
__device__ inline float block_red(float v, float* red) {
    int t = threadIdx.x;
    red[t] = v; __syncthreads();
    for (int s = 128; s > 0; s >>= 1) { if (t < s) red[t] += red[t + s]; __syncthreads(); }
    float r = red[0]; __syncthreads();
    return r;
}

__global__ __launch_bounds__(256) void k_ln(const float* __restrict__ X,
                                            const float* __restrict__ w, const float* __restrict__ b,
                                            float* __restrict__ Y32, __bf16* __restrict__ Yb) {
    __shared__ float red[256];
    int row = blockIdx.x, tid = threadIdx.x;
    const float* x = X + (size_t)row * Cc;
    float s = 0.f;
    for (int i = 0; i < 4; ++i) s += x[tid + i * 256];
    float mean = block_red(s, red) * (1.f / Cc);
    float vs = 0.f;
    for (int i = 0; i < 4; ++i) { float d = x[tid + i * 256] - mean; vs += d * d; }
    float var = block_red(vs, red) * (1.f / Cc);
    float inv = rsqrtf(var + 1e-5f);
    for (int i = 0; i < 4; ++i) {
        int c = tid + i * 256;
        float y = (x[c] - mean) * inv * w[c] + b[c];
        Y32[(size_t)row * Cc + c] = y;
        Yb [(size_t)row * Cc + c] = (__bf16)y;
    }
}

// ---------------- WMMA GEMM: out[M,Nd] = A[M,Kd](bf16) @ W[Nd,Kd]^T ----------------
// Block tile 128x64, 8 waves, each wave owns 32x32 (4 WMMA accumulators).
// A tile (bf16) is moved by the Tensor Data Mover into padded LDS rows;
// W tile (f32) is converted to bf16 on the fly by VALU.
__global__ __launch_bounds__(256) void k_gemm(const __bf16* __restrict__ A,
                                              const float* __restrict__ W,
                                              const float* __restrict__ bias,
                                              const float* __restrict__ resid,
                                              float* __restrict__ outF,
                                              __bf16* __restrict__ outB,
                                              int M, int Nd, int Kd, int relu) {
    __shared__ __bf16 As[128][40];   // row stride 80B (16B aligned; TDM pad 4 dw / 16 dw)
    __shared__ __bf16 Bs[64][40];
    int tid = threadIdx.x;
    int nbn = Nd >> 6;
    int bm = blockIdx.x / nbn, bn = blockIdx.x % nbn;
    int w = tid >> 5, lane = tid & 31;
    int wm = w & 3, wn = w >> 2;
    int lrow = tid >> 2, lcol = (tid & 3) * 8;

    v8f acc[2][2] = {{{}, {}}, {{}, {}}};
    for (int kt = 0; kt < Kd; kt += 32) {
        // ---- W tile 64x32: f32 -> bf16 into LDS ----
        const float* wp = W + (size_t)(bn * 64 + lrow) * Kd + kt + lcol;
        __builtin_prefetch(wp + 32, 0, 1);
        float4 f0 = *(const float4*)(wp);
        float4 f1 = *(const float4*)(wp + 4);
        uint4 bv;
        bv.x = pk_bf16(f0.x, f0.y); bv.y = pk_bf16(f0.z, f0.w);
        bv.z = pk_bf16(f1.x, f1.y); bv.w = pk_bf16(f1.z, f1.w);
        *(uint4*)&Bs[lrow][lcol] = bv;

        // ---- A tile 128x32 bf16 ----
#if USE_TDM
        if (tid < 32) {   // wave 0 issues the DMA; TDM ignores EXEC
            unsigned lds_base = (unsigned)(size_t)(&As[0][0]);
            unsigned long long ga =
                (unsigned long long)(size_t)A + ((size_t)(bm * 128) * Kd + kt) * 2ull;
            v4u g0;
            g0.x = 1u;                                         // count=1 valid descriptor
            g0.y = lds_base;                                   // lds_addr
            g0.z = (unsigned)ga;                               // global_addr[31:0]
            g0.w = (unsigned)((ga >> 32) & 0x01FFFFFFull) | 0x80000000u; // addr[56:32] | type=2
            v8i g1;
            g1[0] = (1 << 16)            // data_size = 2 bytes
                  | (1 << 20)            // pad_enable
                  | (3 << 22)            // pad_interval: 16 DWORDs (one 32-elem bf16 row)
                  | (3 << 25);           // pad_amount: 4 DWORDs (8 bf16 -> stride 40)
            g1[1] = (int)(((unsigned)Kd & 0xFFFFu) << 16);                       // dim0[15:0]
            g1[2] = (int)(((unsigned)Kd >> 16) | (((unsigned)M & 0xFFFFu) << 16)); // dim0 hi | dim1 lo
            g1[3] = (int)(((unsigned)M >> 16) | (32u << 16));                    // dim1 hi | tile_dim0=32
            g1[4] = 128;                                                         // tile_dim1=128
            g1[5] = Kd;                                                          // dim0_stride lo32
            g1[6] = 0; g1[7] = 0;
            v4i gz4 = {0, 0, 0, 0};
            v8i gz8 = {0, 0, 0, 0, 0, 0, 0, 0};
            __builtin_amdgcn_tensor_load_to_lds(g0, g1, gz4, gz4, gz8, 0);
            __builtin_amdgcn_s_wait_tensorcnt(0);
        }
#else
        {
            int arow = tid >> 1, acol = (tid & 1) * 16;
            const uint4* ap = (const uint4*)(A + (size_t)(bm * 128 + arow) * Kd + kt + acol);
            uint4 a0 = ap[0], a1 = ap[1];
            *(uint4*)&As[arow][acol] = a0;
            *(uint4*)&As[arow][acol + 8] = a1;
        }
#endif
        __syncthreads();

        FragB a0, a1, b0, b1;
        int m0 = wm * 32 + (lane & 15);
        int ak = (lane >> 4) * 16;
        a0.u[0] = *(const uint4*)&As[m0][ak];
        a0.u[1] = *(const uint4*)&As[m0][ak + 8];
        a1.u[0] = *(const uint4*)&As[m0 + 16][ak];
        a1.u[1] = *(const uint4*)&As[m0 + 16][ak + 8];
        int n  = lane & 15;
        int bk = (lane >> 4) * 16;
        b0.u[0] = *(const uint4*)&Bs[wn * 32 + n][bk];
        b0.u[1] = *(const uint4*)&Bs[wn * 32 + n][bk + 8];
        b1.u[0] = *(const uint4*)&Bs[wn * 32 + 16 + n][bk];
        b1.u[1] = *(const uint4*)&Bs[wn * 32 + 16 + n][bk + 8];
        acc[0][0] = __builtin_amdgcn_wmma_f32_16x16x32_bf16(false, a0.v, false, b0.v, (short)0, acc[0][0], false, false);
        acc[0][1] = __builtin_amdgcn_wmma_f32_16x16x32_bf16(false, a0.v, false, b1.v, (short)0, acc[0][1], false, false);
        acc[1][0] = __builtin_amdgcn_wmma_f32_16x16x32_bf16(false, a1.v, false, b0.v, (short)0, acc[1][0], false, false);
        acc[1][1] = __builtin_amdgcn_wmma_f32_16x16x32_bf16(false, a1.v, false, b1.v, (short)0, acc[1][1], false, false);
        __syncthreads();
    }
    int rbase = (lane >> 4) * 8;
    int col0  = bn * 64 + wn * 32 + (lane & 15);
    #pragma unroll
    for (int sub = 0; sub < 2; ++sub) {
        #pragma unroll
        for (int half = 0; half < 2; ++half) {
            #pragma unroll
            for (int r = 0; r < 8; ++r) {
                int row = bm * 128 + wm * 32 + sub * 16 + rbase + r;
                int c   = col0 + half * 16;
                float v = acc[sub][half][r];
                if (bias)  v += bias[c];
                if (relu)  v = fmaxf(v, 0.f);
                if (resid) v += resid[(size_t)row * Nd + c];
                if (outF)  outF[(size_t)row * Nd + c] = v;
                if (outB)  outB[(size_t)row * Nd + c] = (__bf16)v;
            }
        }
    }
}

// ---------------- flash attention: one WG per (b, h, 64-query tile) ----------------
__global__ __launch_bounds__(256) void k_attn(const __bf16* __restrict__ QKVB,
                                              __bf16* __restrict__ ATTB) {
    __shared__ __bf16 Qs[64][72];   // row stride 144B
    __shared__ __bf16 Ks[64][72];
    __shared__ __bf16 Vs[64][72];
    __shared__ float  Ss[64][66];
    __shared__ __bf16 Ps[64][72];
    __shared__ float rowm[64], rowl[64], rowa[64];

    int tid = threadIdx.x;
    int qt = blockIdx.x & 15;
    int h  = (blockIdx.x >> 4) & (Hc - 1);
    int b  = blockIdx.x >> 8;
    size_t rowb = (size_t)b * Tc;
    int w = tid >> 5, lane = tid & 31;
    int wm = w & 3, wn = w >> 2;
    int rbase = (lane >> 4) * 8;
    const float scale = 0.125f;   // 1/sqrt(64)

    // load Q tile
    {
        int r = tid >> 2, cc = (tid & 3) * 16;
        const uint4* src = (const uint4*)(QKVB + (rowb + qt * 64 + r) * C3 + h * HDc + cc);
        uint4 a0 = src[0], a1 = src[1];
        *(uint4*)&Qs[r][cc] = a0; *(uint4*)&Qs[r][cc + 8] = a1;
    }
    if (tid < 64) { rowm[tid] = -INFINITY; rowl[tid] = 0.f; }
    v8f o0 = {}, o1 = {};

    for (int kb = 0; kb < 16; ++kb) {
        __syncthreads();
        // load K,V tiles for keys kb*64..+63
        {
            int r = tid >> 2, cc = (tid & 3) * 16;
            const uint4* sk = (const uint4*)(QKVB + (rowb + kb * 64 + r) * C3 + Cc    + h * HDc + cc);
            const uint4* sv = (const uint4*)(QKVB + (rowb + kb * 64 + r) * C3 + 2*Cc  + h * HDc + cc);
            uint4 k0 = sk[0], k1 = sk[1], v0 = sv[0], v1 = sv[1];
            *(uint4*)&Ks[r][cc] = k0; *(uint4*)&Ks[r][cc + 8] = k1;
            *(uint4*)&Vs[r][cc] = v0; *(uint4*)&Vs[r][cc + 8] = v1;
        }
        __syncthreads();

        // S = Q K^T  (wave: 16 q-rows x 32 keys)
        v8f s0 = {}, s1 = {};
        #pragma unroll
        for (int ks = 0; ks < 64; ks += 32) {
            FragB a, b0, b1;
            int m = wm * 16 + (lane & 15);
            int ak = ks + (lane >> 4) * 16;
            a.u[0] = *(const uint4*)&Qs[m][ak];
            a.u[1] = *(const uint4*)&Qs[m][ak + 8];
            int n = lane & 15;
            int bk = ks + (lane >> 4) * 16;
            b0.u[0] = *(const uint4*)&Ks[wn * 32 + n][bk];
            b0.u[1] = *(const uint4*)&Ks[wn * 32 + n][bk + 8];
            b1.u[0] = *(const uint4*)&Ks[wn * 32 + 16 + n][bk];
            b1.u[1] = *(const uint4*)&Ks[wn * 32 + 16 + n][bk + 8];
            s0 = __builtin_amdgcn_wmma_f32_16x16x32_bf16(false, a.v, false, b0.v, (short)0, s0, false, false);
            s1 = __builtin_amdgcn_wmma_f32_16x16x32_bf16(false, a.v, false, b1.v, (short)0, s1, false, false);
        }
        #pragma unroll
        for (int r = 0; r < 8; ++r) {
            Ss[wm * 16 + rbase + r][wn * 32 + (lane & 15)]      = s0[r] * scale;
            Ss[wm * 16 + rbase + r][wn * 32 + 16 + (lane & 15)] = s1[r] * scale;
        }
        __syncthreads();

        // online softmax per row
        if (tid < 64) {
            float mold = rowm[tid], mnew = mold;
            for (int j = 0; j < 64; ++j) mnew = fmaxf(mnew, Ss[tid][j]);
            float alpha = __expf(mold - mnew);
            float sum = 0.f;
            for (int j = 0; j < 64; ++j) {
                float e = __expf(Ss[tid][j] - mnew);
                sum += e;
                Ps[tid][j] = (__bf16)e;
            }
            rowl[tid] = rowl[tid] * alpha + sum;
            rowm[tid] = mnew;
            rowa[tid] = alpha;
        }
        __syncthreads();

        // rescale O, then O += P V
        #pragma unroll
        for (int r = 0; r < 8; ++r) {
            float al = rowa[wm * 16 + rbase + r];
            o0[r] *= al; o1[r] *= al;
        }
        #pragma unroll
        for (int ks = 0; ks < 64; ks += 32) {
            FragB a, b0, b1;
            int m = wm * 16 + (lane & 15);
            int ak = ks + (lane >> 4) * 16;
            a.u[0] = *(const uint4*)&Ps[m][ak];
            a.u[1] = *(const uint4*)&Ps[m][ak + 8];
            int hd0 = wn * 32 + (lane & 15);
            int bk = ks + (lane >> 4) * 16;
            #pragma unroll
            for (int j = 0; j < 16; ++j) {
                b0.v[j] = Vs[bk + j][hd0];
                b1.v[j] = Vs[bk + j][hd0 + 16];
            }
            o0 = __builtin_amdgcn_wmma_f32_16x16x32_bf16(false, a.v, false, b0.v, (short)0, o0, false, false);
            o1 = __builtin_amdgcn_wmma_f32_16x16x32_bf16(false, a.v, false, b1.v, (short)0, o1, false, false);
        }
    }
    // normalize and write
    #pragma unroll
    for (int r = 0; r < 8; ++r) {
        int qrow = wm * 16 + rbase + r;
        float il = 1.f / rowl[qrow];
        size_t t = rowb + qt * 64 + qrow;
        ATTB[t * Cc + h * HDc + wn * 32 + (lane & 15)]      = (__bf16)(o0[r] * il);
        ATTB[t * Cc + h * HDc + wn * 32 + 16 + (lane & 15)] = (__bf16)(o1[r] * il);
    }
}

// ---------------- router logits (clean + noise nets) ----------------
__global__ void k_router(const float* __restrict__ XN,
                         const float* __restrict__ rtw, const float* __restrict__ rtb,
                         const float* __restrict__ nzw, const float* __restrict__ nzb,
                         float* __restrict__ LOG, float* __restrict__ NZ) {
    int g = blockIdx.x * 256 + threadIdx.x;     // Nc*Ec
    int e = g & (Ec - 1), row = g >> 3;
    const float* x = XN + (size_t)row * Cc;
    const float* w1 = rtw + (size_t)e * Cc;
    const float* w2 = nzw + (size_t)e * Cc;
    float s1 = 0.f, s2 = 0.f;
    for (int c = 0; c < Cc; ++c) { float xv = x[c]; s1 += xv * w1[c]; s2 += xv * w2[c]; }
    LOG[g] = s1 + rtb[e];
    NZ[g]  = s2 + nzb[e];
}

// ---------------- noisy top-2 gating ----------------
__global__ void k_topk(const float* __restrict__ LOG, const float* __restrict__ NZ,
                       const float* __restrict__ noise_l,
                       int* __restrict__ TOPI, float* __restrict__ GV) {
    int row = blockIdx.x * 256 + threadIdx.x;
    if (row >= Nc) return;
    float nv[Ec];
    for (int e = 0; e < Ec; ++e) {
        float z = NZ[row * Ec + e];
        float sp = (z > 30.f) ? z : log1pf(__expf(z));
        nv[e] = LOG[row * Ec + e] + noise_l[(size_t)row * Ec + e] * sp;
    }
    int i0 = 0; float v0 = nv[0];
    for (int e = 1; e < Ec; ++e) if (nv[e] > v0) { v0 = nv[e]; i0 = e; }
    int i1 = -1; float v1 = -INFINITY;
    for (int e = 0; e < Ec; ++e) if (e != i0 && nv[e] > v1) { v1 = nv[e]; i1 = e; }
    float e1 = __expf(v1 - v0);
    float g0 = 1.f / (1.f + e1), g1 = e1 / (1.f + e1);
    TOPI[row * 2] = i0; TOPI[row * 2 + 1] = i1;
    GV[row * 2] = g0;   GV[row * 2 + 1] = g1;
}

// ---------------- capacity-limited expert assignment ----------------
__global__ void k_assign(const int* __restrict__ TOPI, const float* __restrict__ GV,
                         int* __restrict__ EIDX, float* __restrict__ EGATE, int* __restrict__ ECNT) {
    int e = threadIdx.x;
    if (e >= Ec) return;
    int cnt = 0;
    for (int t = 0; t < Nc; ++t) {
        int a = TOPI[t * 2], b = TOPI[t * 2 + 1];
        if (a == e || b == e) {
            if (cnt < CAPc) {
                EIDX[e * CAPc + cnt]  = t;
                EGATE[e * CAPc + cnt] = (a == e) ? GV[t * 2] : GV[t * 2 + 1];
                ++cnt;
            }
        }
    }
    ECNT[e] = cnt;
}

__global__ void k_gather(const __bf16* __restrict__ XNB, const int* __restrict__ EIDX,
                         const int* __restrict__ ECNT, __bf16* __restrict__ XIB, int e) {
    int g = blockIdx.x * 256 + threadIdx.x;    // CAPc*Cc
    int j = g >> 10, c = g & (Cc - 1);
    int cnt = ECNT[e];
    XIB[g] = (j < cnt) ? XNB[(size_t)EIDX[e * CAPc + j] * Cc + c] : (__bf16)0.0f;
}

__global__ void k_scatter(const float* __restrict__ EOUT, const int* __restrict__ EIDX,
                          const float* __restrict__ EGATE, const int* __restrict__ ECNT,
                          float* __restrict__ X, int e) {
    int g = blockIdx.x * 256 + threadIdx.x;    // CAPc*Cc
    int j = g >> 10, c = g & (Cc - 1);
    if (j < ECNT[e]) {
        X[(size_t)EIDX[e * CAPc + j] * Cc + c] += EOUT[g] * EGATE[e * CAPc + j];
    }
}

// ---------------- final mean over T and head ----------------
__global__ void k_mean(const float* __restrict__ XN, float* __restrict__ MEAN) {
    int g = blockIdx.x * 256 + threadIdx.x;    // Bc*Cc
    int b = g >> 10, c = g & (Cc - 1);
    float s = 0.f;
    for (int t = 0; t < Tc; ++t) s += XN[((size_t)b * Tc + t) * Cc + c];
    MEAN[g] = s * (1.f / Tc);
}

__global__ __launch_bounds__(256) void k_head(const float* __restrict__ MEAN,
                                              const float* __restrict__ hw, const float* __restrict__ hb,
                                              float* __restrict__ out) {
    __shared__ float red[256];
    int b = blockIdx.x, tid = threadIdx.x;
    float s = 0.f;
    for (int i = 0; i < 4; ++i) { int c = tid + i * 256; s += MEAN[(size_t)b * Cc + c] * hw[c]; }
    float r = block_red(s, red);
    if (tid == 0) out[b] = r + hb[0];
}

// ---------------- host orchestration ----------------
extern "C" void kernel_launch(void* const* d_in, const int* in_sizes, int n_in,
                              void* d_out, int out_size, void* d_ws, size_t ws_size,
                              hipStream_t stream) {
    (void)in_sizes; (void)n_in; (void)out_size; (void)ws_size;
    const int*   ids   = (const int*)  d_in[0];
    const float* noise = (const float*)d_in[1];
    const float* tok   = (const float*)d_in[2];
    const float* pos   = (const float*)d_in[3];
    const float* ln1w  = (const float*)d_in[4];
    const float* ln1b  = (const float*)d_in[5];
    const float* qkvw  = (const float*)d_in[6];
    const float* outw  = (const float*)d_in[7];
    const float* ln2w  = (const float*)d_in[8];
    const float* ln2b  = (const float*)d_in[9];
    const float* rtw   = (const float*)d_in[10];
    const float* rtb   = (const float*)d_in[11];
    const float* nzw   = (const float*)d_in[12];
    const float* nzb   = (const float*)d_in[13];
    const float* ew1   = (const float*)d_in[14];
    const float* eb1   = (const float*)d_in[15];
    const float* ew2   = (const float*)d_in[16];
    const float* eb2   = (const float*)d_in[17];
    const float* lnfw  = (const float*)d_in[18];
    const float* lnfb  = (const float*)d_in[19];
    const float* headw = (const float*)d_in[20];
    const float* headb = (const float*)d_in[21];

    size_t off = 0;
    char* base = (char*)d_ws;
    auto alloc = [&](size_t bytes) -> char* {
        char* r = base + off;
        off = (off + bytes + 255) & ~(size_t)255;
        return r;
    };
    float*  X     = (float*) alloc((size_t)Nc * Cc * 4);
    float*  XN32  = (float*) alloc((size_t)Nc * Cc * 4);
    __bf16* XNB   = (__bf16*)alloc((size_t)Nc * Cc * 2);
    __bf16* QKVB  = (__bf16*)alloc((size_t)Nc * C3 * 2);
    __bf16* ATTB  = (__bf16*)alloc((size_t)Nc * Cc * 2);
    float*  LOG   = (float*) alloc((size_t)Nc * Ec * 4);
    float*  NZb   = (float*) alloc((size_t)Nc * Ec * 4);
    int*    TOPI  = (int*)   alloc((size_t)Nc * 2 * 4);
    float*  GV    = (float*) alloc((size_t)Nc * 2 * 4);
    int*    EIDX  = (int*)   alloc((size_t)Ec * CAPc * 4);
    float*  EGATE = (float*) alloc((size_t)Ec * CAPc * 4);
    int*    ECNT  = (int*)   alloc((size_t)Ec * 4);
    __bf16* XIB   = (__bf16*)alloc((size_t)CAPc * Cc * 2);
    __bf16* HB    = (__bf16*)alloc((size_t)CAPc * C4 * 2);
    float*  EOUT  = (float*) alloc((size_t)CAPc * Cc * 4);
    float*  MEAN  = (float*) alloc((size_t)Bc * Cc * 4);

    k_embed<<<Nc * Cc / 256, 256, 0, stream>>>(ids, tok, pos, X);

    for (int l = 0; l < Lc; ++l) {
        // attention block
        k_ln<<<Nc, 256, 0, stream>>>(X, ln1w + (size_t)l * Cc, ln1b + (size_t)l * Cc, XN32, XNB);
        k_gemm<<<(Nc / 128) * (C3 / 64), 256, 0, stream>>>(
            XNB, qkvw + (size_t)l * C3 * Cc, nullptr, nullptr, nullptr, QKVB, Nc, C3, Cc, 0);
        k_attn<<<Bc * Hc * (Tc / 64), 256, 0, stream>>>(QKVB, ATTB);
        k_gemm<<<(Nc / 128) * (Cc / 64), 256, 0, stream>>>(
            ATTB, outw + (size_t)l * Cc * Cc, nullptr, X, X, nullptr, Nc, Cc, Cc, 0);

        // MoE block
        k_ln<<<Nc, 256, 0, stream>>>(X, ln2w + (size_t)l * Cc, ln2b + (size_t)l * Cc, XN32, XNB);
        k_router<<<Nc * Ec / 256, 256, 0, stream>>>(
            XN32, rtw + (size_t)l * Ec * Cc, rtb + (size_t)l * Ec,
            nzw + (size_t)l * Ec * Cc, nzb + (size_t)l * Ec, LOG, NZb);
        k_topk<<<Nc / 256, 256, 0, stream>>>(LOG, NZb, noise + (size_t)l * Nc * Ec, TOPI, GV);
        k_assign<<<1, 32, 0, stream>>>(TOPI, GV, EIDX, EGATE, ECNT);
        for (int e = 0; e < Ec; ++e) {
            size_t we = (size_t)(l * Ec + e);
            k_gather<<<CAPc * Cc / 256, 256, 0, stream>>>(XNB, EIDX, ECNT, XIB, e);
            k_gemm<<<(CAPc / 128) * (C4 / 64), 256, 0, stream>>>(
                XIB, ew1 + we * C4 * Cc, eb1 + we * C4, nullptr, nullptr, HB, CAPc, C4, Cc, 1);
            k_gemm<<<(CAPc / 128) * (Cc / 64), 256, 0, stream>>>(
                HB, ew2 + we * Cc * C4, eb2 + we * Cc, nullptr, EOUT, nullptr, CAPc, Cc, C4, 0);
            k_scatter<<<CAPc * Cc / 256, 256, 0, stream>>>(EOUT, EIDX, EGATE, ECNT, X, e);
        }
    }

    k_ln<<<Nc, 256, 0, stream>>>(X, lnfw, lnfb, XN32, XNB);
    k_mean<<<Bc * Cc / 256, 256, 0, stream>>>(XN32, MEAN);
    k_head<<<Bc, 256, 0, stream>>>(MEAN, headw, headb, (float*)d_out);
}
